// SwinBlock_75299366633487
// MI455X (gfx1250) — matrix operand order repaired and checked by reading, not compiled, not run on hardware
//
#include <hip/hip_runtime.h>
#include <math.h>

typedef _Float16 half_t;
typedef __attribute__((ext_vector_type(16))) _Float16 v16h;
typedef __attribute__((ext_vector_type(8)))  float    v8f;

#define HEADS_  16
#define HD_     32
#define C_DIM   512
#define FFN_    2048
#define SHIFT_  4
#define M_TOK   65536        // B * H * W = 16*64*64
#define SCALE_  0.17677669529663687f   // 32^-0.5
#define LDK     48           // padded LDS row stride (halves): 96B, keeps b128 16B-aligned

// ---- roll + window-partition index maps (pure permutations) ----
__device__ __forceinline__ int tok_to_win(int tok) {
  int b = tok >> 12;
  int h = (tok >> 6) & 63;
  int w = tok & 63;
  int hp = (h - SHIFT_) & 63;          // position after roll(-SHIFT)
  int wp = (w - SHIFT_) & 63;
  int wi = ((hp >> 3) << 3) + (wp >> 3);
  int ti = ((hp & 7) << 3) + (wp & 7);
  return (b << 12) + (wi << 6) + ti;
}
__device__ __forceinline__ int win_to_tok(int r) {
  int b = r >> 12;
  int wi = (r >> 6) & 63;
  int ti = r & 63;
  int hp = ((wi >> 3) << 3) + (ti >> 3);
  int wp = ((wi & 7) << 3) + (ti & 7);
  int h = (hp + SHIFT_) & 63;          // roll(+SHIFT) on the way back
  int w = (wp + SHIFT_) & 63;
  return (b << 12) + (h << 6) + w;
}
__device__ __forceinline__ int regf(int p) { return p < 56 ? 0 : (p < 60 ? 1 : 2); }

// ---- CDNA5 async global->LDS DMA (16B per lane), tracked by ASYNCcnt ----
__device__ __forceinline__ void async_copy16(unsigned int lds_off, const half_t* g) {
  asm volatile("global_load_async_to_lds_b128 %0, %1, off"
               :: "v"(lds_off), "v"(g) : "memory");
}
__device__ __forceinline__ unsigned int lds_off32(const void* p) {
  // flat LDS aperture keeps the LDS byte offset in addr[31:0] (ISA 10.2)
  return (unsigned int)(unsigned long long)p;
}

// ---- fp32 -> fp16 weight conversion ----
__global__ void cvt_f32_f16(const float* __restrict__ s, half_t* __restrict__ d, int n) {
  int i = blockIdx.x * 256 + threadIdx.x;
  if (i < n) d[i] = (half_t)s[i];
}

// ---- LayerNorm over 512 channels, one wave per token, optional window permute ----
__global__ __launch_bounds__(256) void ln_kernel(
    const float* __restrict__ X, const float* __restrict__ g,
    const float* __restrict__ bb, half_t* __restrict__ O, int permute) {
  const int lane = threadIdx.x & 31, wv = threadIdx.x >> 5;
  const int tok = blockIdx.x * 8 + wv;
  const float4* row = (const float4*)(X + (size_t)tok * C_DIM);
  float4 vx[4];
  float s1 = 0.f, s2 = 0.f;
#pragma unroll
  for (int i = 0; i < 4; ++i) {
    vx[i] = row[i * 32 + lane];
    s1 += vx[i].x + vx[i].y + vx[i].z + vx[i].w;
    s2 += vx[i].x * vx[i].x + vx[i].y * vx[i].y + vx[i].z * vx[i].z + vx[i].w * vx[i].w;
  }
#pragma unroll
  for (int off = 1; off < 32; off <<= 1) {
    s1 += __shfl_xor(s1, off, 32);
    s2 += __shfl_xor(s2, off, 32);
  }
  float mean = s1 * (1.f / 512.f);
  float var  = s2 * (1.f / 512.f) - mean * mean;
  float rs   = rsqrtf(var + 1e-5f);
  int orow = permute ? tok_to_win(tok) : tok;
  half_t* od = O + (size_t)orow * C_DIM;
#pragma unroll
  for (int i = 0; i < 4; ++i) {
    int c = (i * 32 + lane) * 4;
    od[c + 0] = (half_t)((vx[i].x - mean) * rs * g[c + 0] + bb[c + 0]);
    od[c + 1] = (half_t)((vx[i].y - mean) * rs * g[c + 1] + bb[c + 1]);
    od[c + 2] = (half_t)((vx[i].z - mean) * rs * g[c + 2] + bb[c + 2]);
    od[c + 3] = (half_t)((vx[i].w - mean) * rs * g[c + 3] + bb[c + 3]);
  }
}

// ---- Generic WMMA GEMM: out = A[M,K](f16) * W[N,K]^T(f16) + bias, fused epilogues ----
// 128x128 block tile, BK=32, 8 waves each computing 32x64 (8 WMMA/step),
// tiles staged via async global->LDS DMA with double buffering.
// mode 0: split K/V fp16   mode 1: *SCALE -> q fp16   mode 2: proj + permuted residual -> f32
// mode 3: GELU -> fp16     mode 4: + residual -> f32 (final output)
__global__ __launch_bounds__(256) void wmma_gemm_kernel(
    const half_t* __restrict__ A, const half_t* __restrict__ W,
    const float* __restrict__ bias, int K, int N, int mode,
    half_t* __restrict__ oh0, half_t* __restrict__ oh1,
    float* __restrict__ of, const float* __restrict__ res) {
  __shared__ half_t sA[2][128][LDK];
  __shared__ half_t sB[2][128][LDK];
  const int tid = threadIdx.x;
  const int lane = tid & 31, wv = tid >> 5;
  const int hb = lane >> 4, n15 = lane & 15;
  const int wr = wv & 3, wc = wv >> 2;       // wave -> 32-row subtile, 64-col group
  const int m0 = blockIdx.x * 128, n0 = blockIdx.y * 128;
  const int lrow = tid >> 2;                 // 0..63 (plus +64 for second chunk)
  const int lch  = (tid & 3) * 8;            // half offset within a 32-half row

  const unsigned int aoff0 = lds_off32(&sA[0][lrow][lch]);
  const unsigned int aoff1 = lds_off32(&sA[0][lrow + 64][lch]);
  const unsigned int boff0 = lds_off32(&sB[0][lrow][lch]);
  const unsigned int boff1 = lds_off32(&sB[0][lrow + 64][lch]);
  const unsigned int bufstep = lds_off32(&sA[1][0][0]) - lds_off32(&sA[0][0][0]);

  v8f acc[2][4];
#pragma unroll
  for (int mt = 0; mt < 2; ++mt)
#pragma unroll
    for (int ct = 0; ct < 4; ++ct) acc[mt][ct] = (v8f){};

  const int nk = K >> 5;
  // prologue: fill buffer 0
  {
    const half_t* ga0 = A + (size_t)(m0 + lrow) * K + lch;
    const half_t* gb0 = W + (size_t)(n0 + lrow) * K + lch;
    async_copy16(aoff0, ga0);
    async_copy16(aoff1, ga0 + (size_t)64 * K);
    async_copy16(boff0, gb0);
    async_copy16(boff1, gb0 + (size_t)64 * K);
  }
  for (int kt = 0; kt < nk; ++kt) {
    const unsigned int cur = (kt & 1) ? bufstep : 0u;
    if (kt + 1 < nk) {
      const int k0n = (kt + 1) << 5;
      const unsigned int nxt = ((kt + 1) & 1) ? bufstep : 0u;
      const half_t* ga = A + (size_t)(m0 + lrow) * K + k0n + lch;
      const half_t* gb = W + (size_t)(n0 + lrow) * K + k0n + lch;
      async_copy16(aoff0 + nxt, ga);
      async_copy16(aoff1 + nxt, ga + (size_t)64 * K);
      async_copy16(boff0 + nxt, gb);
      async_copy16(boff1 + nxt, gb + (size_t)64 * K);
      asm volatile("s_wait_asynccnt 0x4" ::: "memory");  // current buffer's 4 loads done
    } else {
      asm volatile("s_wait_asynccnt 0x0" ::: "memory");
    }
    __syncthreads();   // all threads' current-buffer DMA complete

    const half_t* bufA = &sA[0][0][0] + (cur >> 1);  // cur is a byte offset; halves = /2
    const half_t* bufB = &sB[0][0][0] + (cur >> 1);

    v16h af[2];
#pragma unroll
    for (int mt = 0; mt < 2; ++mt) {
      const half_t* p = bufA + (wr * 32 + mt * 16 + n15) * LDK + hb * 8;
#pragma unroll
      for (int e = 0; e < 8; ++e) { af[mt][e] = p[e]; af[mt][e + 8] = p[e + 16]; }
    }
    v16h bf[4];
#pragma unroll
    for (int ct = 0; ct < 4; ++ct) {
      const half_t* p = bufB + (wc * 64 + ct * 16 + n15) * LDK + hb * 8;
#pragma unroll
      for (int e = 0; e < 16; ++e) bf[ct][e] = p[e + (e & 8)];
    }
#pragma unroll
    for (int mt = 0; mt < 2; ++mt)
#pragma unroll
      for (int ct = 0; ct < 4; ++ct)
        acc[mt][ct] = __builtin_amdgcn_wmma_f32_16x16x32_f16(
            false, af[mt], false, bf[ct], (short)0, acc[mt][ct], false, false);

    __syncthreads();   // everyone done reading before this buffer is refilled
  }

#pragma unroll
  for (int mt = 0; mt < 2; ++mt) {
#pragma unroll
    for (int ct = 0; ct < 4; ++ct) {
      int cg = n0 + wc * 64 + ct * 16 + n15;
      float bv = bias[cg];
#pragma unroll
      for (int j = 0; j < 8; ++j) {
        int r = m0 + wr * 32 + mt * 16 + hb * 8 + j;
        float val = acc[mt][ct][j] + bv;
        if (mode == 0) {                         // qkv: split into K and V
          if (cg < C_DIM) oh0[(size_t)r * C_DIM + cg] = (half_t)val;
          else            oh1[(size_t)r * C_DIM + (cg - C_DIM)] = (half_t)val;
        } else if (mode == 1) {                  // q projection, fold in SCALE
          oh0[(size_t)r * C_DIM + cg] = (half_t)(val * SCALE_);
        } else if (mode == 2) {                  // proj + window-reverse+roll + residual
          int t = win_to_tok(r);
          of[(size_t)t * C_DIM + cg] = res[(size_t)t * C_DIM + cg] + val;
        } else if (mode == 3) {                  // fc1 + exact GELU
          float gg = 0.5f * val * (1.0f + erff(val * 0.70710678118654752f));
          oh0[(size_t)r * (size_t)N + cg] = (half_t)gg;
        } else {                                 // fc2 + residual -> final output
          of[(size_t)r * C_DIM + cg] = res[(size_t)r * C_DIM + cg] + val;
        }
      }
    }
  }
}

// ---- windowed attention: one block per window, one wave per head (x2 heads) ----
__global__ __launch_bounds__(256) void attn_kernel(
    const half_t* __restrict__ Q, const half_t* __restrict__ Km,
    const half_t* __restrict__ Vm, const float* __restrict__ relb,
    half_t* __restrict__ AO) {
  __shared__ half_t p_lds[8][64 * 64];       // per-wave softmax(P) staging, 64KB
  const int lane = threadIdx.x & 31, wv = threadIdx.x >> 5;
  const int hb = lane >> 4, n15 = lane & 15;
  const int rb = blockIdx.x * 64;            // window base row
  const int win = blockIdx.x & 63;           // window index within image
  const int wh = win >> 3, ww = win & 7;

  for (int hi = 0; hi < 2; ++hi) {
    const int head = wv + hi * 8;
    const half_t* q = Q  + (size_t)rb * C_DIM + head * HD_;
    const half_t* k = Km + (size_t)rb * C_DIM + head * HD_;
    const half_t* v = Vm + (size_t)rb * C_DIM + head * HD_;

    // S = q @ k^T : 4x4 tiles of 16x16, K=32 == one WMMA each
    v16h qf[4], kf[4];
#pragma unroll
    for (int rt = 0; rt < 4; ++rt) {
      const half_t* p = q + (size_t)(rt * 16 + n15) * C_DIM + hb * 8;
#pragma unroll
      for (int e = 0; e < 8; ++e) { qf[rt][e] = p[e]; qf[rt][e + 8] = p[e + 16]; }
    }
#pragma unroll
    for (int ctt = 0; ctt < 4; ++ctt) {
      const half_t* p = k + (size_t)(ctt * 16 + n15) * C_DIM + hb * 8;
#pragma unroll
      for (int e = 0; e < 16; ++e) kf[ctt][e] = p[e + (e & 8)];
    }
    v8f S[4][4];
#pragma unroll
    for (int rt = 0; rt < 4; ++rt)
#pragma unroll
      for (int ct = 0; ct < 4; ++ct) {
        v8f z = {};
        S[rt][ct] = __builtin_amdgcn_wmma_f32_16x16x32_f16(false, qf[rt], false, kf[ct], (short)0, z, false, false);
      }

    // + relative position bias + shift mask
#pragma unroll
    for (int ct = 0; ct < 4; ++ct) {
      int jc = ct * 16 + n15;
      int jr = 63 - jc;
      int cj = (jr >> 3) * 15 + (jr & 7);
      int regj = regf(wh * 8 + (jc >> 3)) * 3 + regf(ww * 8 + (jc & 7));
#pragma unroll
      for (int rt = 0; rt < 4; ++rt)
#pragma unroll
        for (int j = 0; j < 8; ++j) {
          int i = rt * 16 + hb * 8 + j;
          int ci = (i >> 3) * 15 + (i & 7);
          int regi = regf(wh * 8 + (i >> 3)) * 3 + regf(ww * 8 + (i & 7));
          float mk = (regi == regj) ? 0.f : -100.f;
          S[rt][ct][j] += relb[(ci + cj) * HEADS_ + head] + mk;
        }
    }

    // row softmax (row spread over 16-lane half-group x 4 column tiles)
#pragma unroll
    for (int rt = 0; rt < 4; ++rt)
#pragma unroll
      for (int j = 0; j < 8; ++j) {
        float mx = fmaxf(fmaxf(S[rt][0][j], S[rt][1][j]), fmaxf(S[rt][2][j], S[rt][3][j]));
#pragma unroll
        for (int off = 1; off < 16; off <<= 1) mx = fmaxf(mx, __shfl_xor(mx, off, 32));
        float sm = 0.f;
#pragma unroll
        for (int ct = 0; ct < 4; ++ct) {
          float e = expf(S[rt][ct][j] - mx);
          S[rt][ct][j] = e; sm += e;
        }
#pragma unroll
        for (int off = 1; off < 16; off <<= 1) sm += __shfl_xor(sm, off, 32);
        float inv = 1.f / sm;
#pragma unroll
        for (int ct = 0; ct < 4; ++ct) S[rt][ct][j] *= inv;
      }

    // stage P in LDS to reshape accumulator-layout -> A-fragment layout
    half_t* pl = p_lds[wv];
#pragma unroll
    for (int rt = 0; rt < 4; ++rt)
#pragma unroll
      for (int ct = 0; ct < 4; ++ct)
#pragma unroll
        for (int j = 0; j < 8; ++j)
          pl[(rt * 16 + hb * 8 + j) * 64 + ct * 16 + n15] = (half_t)S[rt][ct][j];

    // O = P @ V : 4x2 tiles, K=64 = two chained WMMAs
#pragma unroll
    for (int rt = 0; rt < 4; ++rt) {
#pragma unroll
      for (int dt = 0; dt < 2; ++dt) {
        v8f o = {};
#pragma unroll
        for (int kk = 0; kk < 2; ++kk) {
          v16h pf;
          const half_t* pp = pl + (size_t)(rt * 16 + n15) * 64 + kk * 32 + hb * 8;
#pragma unroll
          for (int e = 0; e < 8; ++e) { pf[e] = pp[e]; pf[e + 8] = pp[e + 16]; }
          v16h vf;
#pragma unroll
          for (int e = 0; e < 16; ++e)
            vf[e] = v[(size_t)(kk * 32 + hb * 8 + e + (e & 8)) * C_DIM + dt * 16 + n15];
          o = __builtin_amdgcn_wmma_f32_16x16x32_f16(false, pf, false, vf, (short)0, o, false, false);
        }
#pragma unroll
        for (int j = 0; j < 8; ++j) {
          int r = rb + rt * 16 + hb * 8 + j;
          AO[(size_t)r * C_DIM + head * HD_ + dt * 16 + n15] = (half_t)o[j];
        }
      }
    }
  }
}

extern "C" void kernel_launch(void* const* d_in, const int* in_sizes, int n_in,
                              void* d_out, int out_size, void* d_ws, size_t ws_size,
                              hipStream_t stream) {
  (void)in_sizes; (void)n_in; (void)out_size; (void)ws_size;
  const float* x      = (const float*)d_in[0];
  const float* skip   = (const float*)d_in[1];
  const float* n1g    = (const float*)d_in[2];
  const float* n1b    = (const float*)d_in[3];
  const float* w_qkv  = (const float*)d_in[4];
  const float* b_qkv  = (const float*)d_in[5];
  const float* w_skip = (const float*)d_in[6];
  const float* b_skip = (const float*)d_in[7];
  const float* relb   = (const float*)d_in[8];
  const float* w_proj = (const float*)d_in[9];
  const float* b_proj = (const float*)d_in[10];
  const float* n2g    = (const float*)d_in[11];
  const float* n2b    = (const float*)d_in[12];
  const float* w_fc1  = (const float*)d_in[13];
  const float* b_fc1  = (const float*)d_in[14];
  const float* w_fc2  = (const float*)d_in[15];
  const float* b_fc2  = (const float*)d_in[16];

  char* ws = (char*)d_ws;
  size_t off = 0;
  auto carve = [&](size_t bytes) -> char* {
    char* p = ws + off;
    off += (bytes + 255) & ~(size_t)255;
    return p;
  };
  half_t* wqkv_h  = (half_t*)carve((size_t)1024 * 512 * 2);
  half_t* wskip_h = (half_t*)carve((size_t)512 * 512 * 2);
  half_t* wproj_h = (half_t*)carve((size_t)512 * 512 * 2);
  half_t* wfc1_h  = (half_t*)carve((size_t)2048 * 512 * 2);
  half_t* wfc2_h  = (half_t*)carve((size_t)512 * 2048 * 2);
  float*  xres    = (float*)carve((size_t)M_TOK * 512 * 4);
  half_t* sn      = (half_t*)carve((size_t)M_TOK * 512 * 2); // LN1(skip), later LN2(x)
  half_t* xn      = (half_t*)carve((size_t)M_TOK * 512 * 2); // LN1(x), later attn-out, later fc1-act span
  half_t* kh      = (half_t*)carve((size_t)M_TOK * 512 * 2);
  half_t* vh      = (half_t*)carve((size_t)M_TOK * 512 * 2);
  half_t* qh      = (half_t*)carve((size_t)M_TOK * 512 * 2);
  half_t* ao = xn;   // attn output reuses xn (dead after qkv GEMM)
  half_t* h1 = xn;   // fc1 activations span xn+kh+vh+qh (256MB, all dead by then)

  // weight conversions (weights stay L2-resident as fp16)
  cvt_f32_f16<<<(1024 * 512 + 255) / 256, 256, 0, stream>>>(w_qkv, wqkv_h, 1024 * 512);
  cvt_f32_f16<<<(512 * 512 + 255) / 256, 256, 0, stream>>>(w_skip, wskip_h, 512 * 512);
  cvt_f32_f16<<<(512 * 512 + 255) / 256, 256, 0, stream>>>(w_proj, wproj_h, 512 * 512);
  cvt_f32_f16<<<(2048 * 512 + 255) / 256, 256, 0, stream>>>(w_fc1, wfc1_h, 2048 * 512);
  cvt_f32_f16<<<(512 * 2048 + 255) / 256, 256, 0, stream>>>(w_fc2, wfc2_h, 512 * 2048);

  // LN1 with fused roll + window-partition permutation
  ln_kernel<<<M_TOK / 8, 256, 0, stream>>>(x,    n1g, n1b, xn, 1);
  ln_kernel<<<M_TOK / 8, 256, 0, stream>>>(skip, n1g, n1b, sn, 1);

  // qkv and skip-q projections
  wmma_gemm_kernel<<<dim3(M_TOK / 128, 1024 / 128), 256, 0, stream>>>(
      xn, wqkv_h, b_qkv, 512, 1024, 0, kh, vh, nullptr, nullptr);
  wmma_gemm_kernel<<<dim3(M_TOK / 128, 512 / 128), 256, 0, stream>>>(
      sn, wskip_h, b_skip, 512, 512, 1, qh, nullptr, nullptr, nullptr);

  // windowed attention (QK^T + rel-bias + shift-mask + softmax + PV)
  attn_kernel<<<1024, 256, 0, stream>>>(qh, kh, vh, relb, ao);

  // output projection with fused window-reverse + roll + residual add
  wmma_gemm_kernel<<<dim3(M_TOK / 128, 512 / 128), 256, 0, stream>>>(
      ao, wproj_h, b_proj, 512, 512, 2, nullptr, nullptr, xres, x);

  // LN2 + FFN
  ln_kernel<<<M_TOK / 8, 256, 0, stream>>>(xres, n2g, n2b, sn, 0);
  wmma_gemm_kernel<<<dim3(M_TOK / 128, 2048 / 128), 256, 0, stream>>>(
      sn, wfc1_h, b_fc1, 512, 2048, 3, h1, nullptr, nullptr, nullptr);
  wmma_gemm_kernel<<<dim3(M_TOK / 128, 512 / 128), 256, 0, stream>>>(
      h1, wfc2_h, b_fc2, 2048, 512, 4, nullptr, nullptr, (float*)d_out, xres);
}
